// NeRTModel_60997125538301
// MI455X (gfx1250) — matrix operand order repaired and checked by reading, not compiled, not run on hardware
//
#include <hip/hip_runtime.h>
#include <hip/hip_bf16.h>

// Problem constants
#define BB   16
#define TT   1024
#define SS   20
#define EE   400
#define HH   512
#define HDIM 256
#define KP1  416     // E padded to multiple of 32
#define GG   1024    // 4*HD
#define H2   1024    // 2*H
#define NPM  448     // mlp_w N (400) padded to multiple of 64
#define BT   (BB*TT) // 16384

typedef __attribute__((ext_vector_type(16))) __bf16       v16bf;
typedef __attribute__((ext_vector_type(8)))  float        v8f;
typedef __attribute__((ext_vector_type(4)))  unsigned int u32x4;

union FragU { u32x4 u[2]; v16bf v; };

// A-matrix (16x32 bf16) fragment: lane m=lane&15, contiguous 16 K values starting at ko=(lane&16)?16:0
__device__ __forceinline__ v16bf load_fragA(const __bf16* row, int lane) {
    int ko = (lane & 16) ? 16 : 0;
    FragU f;
    f.u[0] = *(const u32x4*)(row + ko);
    f.u[1] = *(const u32x4*)(row + ko + 8);
    return f.v;
}

// B-matrix (32x16 bf16) fragment from N x K row-major weight: lane n=lane&15,
// contiguous 16 K values starting at (lane&16)?16:0
__device__ __forceinline__ v16bf load_fragB(const __bf16* row, int lane) {
    int kb = (lane & 16) ? 16 : 0;
    FragU f;
    f.u[0] = *(const u32x4*)(row + kb);
    f.u[1] = *(const u32x4*)(row + kb + 8);
    return f.v;
}

__device__ __forceinline__ float sigm(float x) { return 1.0f / (1.0f + __expf(-x)); }

// Async global->LDS copy of 16 bytes per lane (ASYNCcnt-tracked, CDNA5 TDM-lite path)
__device__ __forceinline__ void async_copy_b128(unsigned lds_byte_addr, const void* gptr) {
    unsigned long long ga = (unsigned long long)gptr;
    asm volatile("global_load_async_to_lds_b128 %0, %1, off"
                 :: "v"(lds_byte_addr), "v"(ga) : "memory");
}
#define WAIT_ASYNC_LE1() asm volatile("s_wait_asynccnt 0x1" ::: "memory")
#define WAIT_ASYNC_0()   asm volatile("s_wait_asynccnt 0x0" ::: "memory")

// ---------------------------------------------------------------------------
// Convert f32 (N x K) weight to zero-padded bf16 (Np x Kp)
// ---------------------------------------------------------------------------
__global__ void pad_convert(const float* __restrict__ src, __bf16* __restrict__ dst,
                            int N, int K, int Np, int Kp) {
    size_t idx = (size_t)blockIdx.x * blockDim.x + threadIdx.x;
    size_t total = (size_t)Np * Kp;
    if (idx >= total) return;
    int n = (int)(idx / Kp);
    int k = (int)(idx % Kp);
    float v = (n < N && k < K) ? src[(size_t)n * K + k] : 0.0f;
    dst[idx] = (__bf16)v;
}

// ---------------------------------------------------------------------------
// x1 -> bf16 padded, and x1 reversed per seq_lens -> bf16 padded
// ---------------------------------------------------------------------------
__global__ void prep_x1(const float* __restrict__ x1, const int* __restrict__ lens,
                        __bf16* __restrict__ x1p, __bf16* __restrict__ x1rp) {
    size_t idx = (size_t)blockIdx.x * blockDim.x + threadIdx.x;
    if (idx >= (size_t)BT * KP1) return;
    int row = (int)(idx / KP1);
    int e   = (int)(idx % KP1);
    int b = row >> 10;          // T = 1024
    int t = row & (TT - 1);
    float v = (e < EE) ? x1[(size_t)row * EE + e] : 0.0f;
    x1p[idx] = (__bf16)v;
    int len = lens[b];
    int rt = (t < len) ? (len - 1 - t) : t;
    float vr = (e < EE) ? x1[((size_t)b * TT + rt) * EE + e] : 0.0f;
    x1rp[idx] = (__bf16)vr;
}

// ---------------------------------------------------------------------------
// Attention collapses to uniform weights: x2ctx = mean over S of x2 (bf16, padded)
// (softmax rows sum to 1 => attn.mean over heads/keys is the constant 1/S)
// ---------------------------------------------------------------------------
__global__ void x2_mean(const float* __restrict__ x2, __bf16* __restrict__ ctx) {
    int row = blockIdx.x;                 // b*T + t
    for (int e = threadIdx.x; e < KP1; e += blockDim.x) {
        float acc = 0.0f;
        if (e < EE) {
            const float* p = x2 + ((size_t)row * SS) * EE + e;
            #pragma unroll
            for (int s = 0; s < SS; ++s) acc += p[(size_t)s * EE];
            acc *= (1.0f / (float)SS);
        }
        ctx[(size_t)row * KP1 + e] = (__bf16)acc;
    }
}

// ---------------------------------------------------------------------------
// Generic bf16 WMMA GEMM: out[m,n] = sum_k A[m,k]*W[n,k] + bias1[n] (+bias2[n])
// A: M x lda bf16 row-major, W: (padded N) x ldw bf16 row-major.
// Block tile 128x64, 256 threads (8 waves), each wave 64x16 (4 accumulators).
// A tile staged via double-buffered async global->LDS DMA (ASYNCcnt).
// ---------------------------------------------------------------------------
__global__ __launch_bounds__(256)
void gemm_bf16_wmma(const __bf16* __restrict__ A, int lda,
                    const __bf16* __restrict__ W, int ldw,
                    const float* __restrict__ bias1, const float* __restrict__ bias2,
                    float* __restrict__ outF, __bf16* __restrict__ outH,
                    int ldo, int col_off,
                    int M, int N, int Kp,
                    const int* __restrict__ seq_lens) {
    __shared__ __bf16 As[2][128 * 32];   // 16 KB double-buffered A tile

    const int tid  = threadIdx.x;
    const int lane = tid & 31;
    const int wave = tid >> 5;
    const int l15  = lane & 15;

    const int row0 = blockIdx.x * 128;
    const int col0 = blockIdx.y * 64;

    const int n_tile = wave & 3;          // 0..3
    const int m_base = (wave >> 2) * 64;  // 0 or 64

    v8f acc[4];
    #pragma unroll
    for (int i = 0; i < 4; ++i) acc[i] = (v8f){0,0,0,0,0,0,0,0};

    const int ldr = tid >> 1;        // A-tile row this thread stages
    const int ldc = (tid & 1) * 16;  // 16-half chunk within the row

    const __bf16* a_src = A + (size_t)(row0 + ldr) * lda + ldc;
    const unsigned lds0 = (unsigned)(unsigned long long)&As[0][ldr * 32 + ldc];
    const unsigned lds1 = (unsigned)(unsigned long long)&As[1][ldr * 32 + ldc];

    const int ksteps = Kp >> 5;

    // Prologue: DMA tile 0 into buffer 0
    async_copy_b128(lds0, a_src);

    const __bf16* wrow = W + (size_t)(col0 + n_tile * 16 + l15) * ldw;

    for (int kt = 0; kt < ksteps; ++kt) {
        const int cur = kt & 1;
        if (kt + 1 < ksteps) {
            // issue next tile's DMA, then wait for the current one (in-order completion)
            async_copy_b128(cur ? lds0 : lds1, a_src + ((kt + 1) << 5));
            WAIT_ASYNC_LE1();
            __builtin_prefetch(wrow + ((kt + 1) << 5), 0, 1);
        } else {
            WAIT_ASYNC_0();
        }
        __syncthreads();

        v16bf bfr = load_fragB(wrow + (kt << 5), lane);
        #pragma unroll
        for (int mt = 0; mt < 4; ++mt) {
            v16bf afr = load_fragA(&As[cur][(m_base + mt * 16 + l15) * 32], lane);
            acc[mt] = __builtin_amdgcn_wmma_f32_16x16x32_bf16(
                false, afr, false, bfr, (short)0, acc[mt], false, false);
        }
        __syncthreads();   // protect buffer reuse before next DMA overwrites it
    }

    const int gcol = col0 + n_tile * 16 + l15;
    if (gcol < N) {
        float bv = bias1 ? bias1[gcol] : 0.0f;
        if (bias2) bv += bias2[gcol];
        #pragma unroll
        for (int mt = 0; mt < 4; ++mt) {
            #pragma unroll
            for (int r = 0; r < 8; ++r) {
                int grow = row0 + m_base + mt * 16 + r + 8 * (lane >> 4);
                float v = acc[mt][r] + bv;
                if (seq_lens) {
                    int b = grow >> 10, t = grow & (TT - 1);
                    if (t >= seq_lens[b]) v = 0.0f;
                }
                size_t o = (size_t)grow * ldo + col_off + gcol;
                if (outF) outF[o] = v; else outH[o] = (__bf16)v;
            }
        }
    }
}

// ---------------------------------------------------------------------------
// LSTM recurrence, batched over B=16 with WMMA: per step (16x256)h @ (256x1024)Whh^T.
// One workgroup (8 waves) per direction; h (bf16), c, gate buffer live in LDS.
// Backward direction stores un-reversed via the ridx involution.
// ---------------------------------------------------------------------------
__global__ __launch_bounds__(256)
void lstm_recurrence(const float* __restrict__ xg_f, const float* __restrict__ xg_b,
                     const __bf16* __restrict__ whh_f, const __bf16* __restrict__ whh_b,
                     float* __restrict__ h_out,  // (B*T) x (2*HD)
                     const int* __restrict__ seq_lens) {
    __shared__ __bf16 h_bf[BB * HDIM];    // 8 KB
    __shared__ float  cst[BB * HDIM];     // 16 KB
    __shared__ float  gbuf[BB * GG];      // 64 KB

    const int dir = blockIdx.x;
    const float*  xg  = dir ? xg_b  : xg_f;
    const __bf16* whh = dir ? whh_b : whh_f;

    const int tid  = threadIdx.x;
    const int lane = tid & 31;
    const int wave = tid >> 5;
    const int l15  = lane & 15;

    for (int i = tid; i < BB * HDIM; i += 256) { h_bf[i] = (__bf16)0.0f; cst[i] = 0.0f; }
    __syncthreads();

    for (int t = 0; t < TT; ++t) {
        // ---- g = h @ Whh^T via WMMA; wave w covers N tiles [w*8, w*8+8) ----
        #pragma unroll
        for (int nti = 0; nti < 8; ++nti) {
            const int nt = wave * 8 + nti;                 // 0..63
            v8f acc = (v8f){0,0,0,0,0,0,0,0};
            #pragma unroll
            for (int kt = 0; kt < 8; ++kt) {
                v16bf afr = load_fragA(&h_bf[l15 * HDIM + (kt << 5)], lane);
                v16bf bfr = load_fragB(whh + (size_t)(nt * 16 + l15) * HDIM + (kt << 5), lane);
                acc = __builtin_amdgcn_wmma_f32_16x16x32_bf16(
                    false, afr, false, bfr, (short)0, acc, false, false);
            }
            const int col = nt * 16 + l15;
            #pragma unroll
            for (int r = 0; r < 8; ++r) {
                int m = r + 8 * (lane >> 4);               // batch index
                gbuf[m * GG + col] = acc[r];
            }
        }
        __syncthreads();

        // ---- elementwise gate update: thread = hidden unit j ----
        const int j = tid;
        #pragma unroll 4
        for (int b = 0; b < BB; ++b) {
            const size_t xrow = ((size_t)b * TT + t) * GG;
            float gi = gbuf[b * GG + j          ] + xg[xrow + j          ];
            float gf = gbuf[b * GG + HDIM + j   ] + xg[xrow + HDIM + j   ];
            float gg = gbuf[b * GG + 2*HDIM + j ] + xg[xrow + 2*HDIM + j ];
            float go = gbuf[b * GG + 3*HDIM + j ] + xg[xrow + 3*HDIM + j ];
            float c = sigm(gf) * cst[b * HDIM + j] + sigm(gi) * tanhf(gg);
            float h = sigm(go) * tanhf(c);
            cst[b * HDIM + j] = c;
            h_bf[b * HDIM + j] = (__bf16)h;
            if (dir == 0) {
                h_out[((size_t)b * TT + t) * HH + j] = h;
            } else {
                int len = seq_lens[b];
                int rt = (t < len) ? (len - 1 - t) : t;
                h_out[((size_t)b * TT + rt) * HH + HDIM + j] = h;
            }
        }
        __syncthreads();
    }
}

// ---------------------------------------------------------------------------
// Masked running mean over T; writes bf16 into fused[:, 0:512]
// ---------------------------------------------------------------------------
__global__ __launch_bounds__(256)
void cumsum_mean(const float* __restrict__ h_out, __bf16* __restrict__ fused,
                 const int* __restrict__ lens) {
    const int b = blockIdx.x >> 1;
    const int j = ((blockIdx.x & 1) << 8) + threadIdx.x;   // 0..511
    const int len = lens[b];
    float acc = 0.0f;
    for (int t = 0; t < TT; ++t) {
        float hv = h_out[((size_t)b * TT + t) * HH + j];
        if (t < len) acc += hv;
        fused[((size_t)b * TT + t) * H2 + j] = (__bf16)(acc / (float)(t + 1));
    }
}

// ---------------------------------------------------------------------------
// Host-side orchestration
// ---------------------------------------------------------------------------
extern "C" void kernel_launch(void* const* d_in, const int* in_sizes, int n_in,
                              void* d_out, int out_size, void* d_ws, size_t ws_size,
                              hipStream_t stream) {
    const float* x1        = (const float*)d_in[0];
    const float* x2        = (const float*)d_in[1];
    const int*   seq_lens  = (const int*)d_in[2];
    // d_in[3] (mha_w) and d_in[4] (mha_b) are mathematically dead: softmax rows sum to 1.
    const float* mlp_mha_w = (const float*)d_in[5];
    const float* mlp_mha_b = (const float*)d_in[6];
    const float* w_ih_f    = (const float*)d_in[7];
    const float* w_hh_f    = (const float*)d_in[8];
    const float* b_ih_f    = (const float*)d_in[9];
    const float* b_hh_f    = (const float*)d_in[10];
    const float* w_ih_b    = (const float*)d_in[11];
    const float* w_hh_b    = (const float*)d_in[12];
    const float* b_ih_b    = (const float*)d_in[13];
    const float* b_hh_b    = (const float*)d_in[14];
    const float* mlp_w     = (const float*)d_in[15];
    const float* mlp_b     = (const float*)d_in[16];
    float* out = (float*)d_out;

    // ---- workspace carve-out ----
    char* base = (char*)d_ws;
    size_t off = 0;
    auto carve = [&](size_t bytes) -> char* {
        char* p = base + off;
        off = (off + bytes + 255) & ~(size_t)255;
        return p;
    };
    __bf16* x1_p     = (__bf16*)carve((size_t)BT * KP1 * 2);
    __bf16* x1r_p    = (__bf16*)carve((size_t)BT * KP1 * 2);
    __bf16* ctx_p    = (__bf16*)carve((size_t)BT * KP1 * 2);
    __bf16* wihf_p   = (__bf16*)carve((size_t)GG * KP1 * 2);
    __bf16* wihb_p   = (__bf16*)carve((size_t)GG * KP1 * 2);
    __bf16* whhf_p   = (__bf16*)carve((size_t)GG * HDIM * 2);
    __bf16* whhb_p   = (__bf16*)carve((size_t)GG * HDIM * 2);
    __bf16* wmha_p   = (__bf16*)carve((size_t)HH * KP1 * 2);
    __bf16* wmlp_p   = (__bf16*)carve((size_t)NPM * H2 * 2);
    float*  xg_f     = (float*)carve((size_t)BT * GG * 4);
    float*  xg_b     = (float*)carve((size_t)BT * GG * 4);
    float*  h_buf    = (float*)carve((size_t)BT * HH * 4);
    __bf16* fused    = (__bf16*)carve((size_t)BT * H2 * 2);

    auto blocks = [](size_t total) { return (unsigned)((total + 255) / 256); };

    // 1) weight conversion / padding to bf16
    pad_convert<<<blocks((size_t)GG * KP1),  256, 0, stream>>>(w_ih_f,    wihf_p, GG, EE,   GG,  KP1);
    pad_convert<<<blocks((size_t)GG * KP1),  256, 0, stream>>>(w_ih_b,    wihb_p, GG, EE,   GG,  KP1);
    pad_convert<<<blocks((size_t)GG * HDIM), 256, 0, stream>>>(w_hh_f,    whhf_p, GG, HDIM, GG,  HDIM);
    pad_convert<<<blocks((size_t)GG * HDIM), 256, 0, stream>>>(w_hh_b,    whhb_p, GG, HDIM, GG,  HDIM);
    pad_convert<<<blocks((size_t)HH * KP1),  256, 0, stream>>>(mlp_mha_w, wmha_p, HH, EE,   HH,  KP1);
    pad_convert<<<blocks((size_t)NPM * H2),  256, 0, stream>>>(mlp_w,     wmlp_p, EE, H2,   NPM, H2);

    // 2) activations
    prep_x1<<<blocks((size_t)BT * KP1), 256, 0, stream>>>(x1, seq_lens, x1_p, x1r_p);
    x2_mean<<<BT, 256, 0, stream>>>(x2, ctx_p);

    // 3) big WMMA GEMMs: LSTM input projections and x2h (bf16 epilogue into fused[:,512:])
    gemm_bf16_wmma<<<dim3(BT/128, GG/64), 256, 0, stream>>>(
        x1_p, KP1, wihf_p, KP1, b_ih_f, b_hh_f, xg_f, nullptr, GG, 0, BT, GG, KP1, nullptr);
    gemm_bf16_wmma<<<dim3(BT/128, GG/64), 256, 0, stream>>>(
        x1r_p, KP1, wihb_p, KP1, b_ih_b, b_hh_b, xg_b, nullptr, GG, 0, BT, GG, KP1, nullptr);
    gemm_bf16_wmma<<<dim3(BT/128, HH/64), 256, 0, stream>>>(
        ctx_p, KP1, wmha_p, KP1, mlp_mha_b, nullptr, nullptr, fused, H2, HH, BT, HH, KP1, nullptr);

    // 4) sequential bidirectional LSTM recurrence (WMMA, LDS-resident state)
    lstm_recurrence<<<2, 256, 0, stream>>>(xg_f, xg_b, whhf_p, whhb_p, h_buf, seq_lens);

    // 5) masked running mean -> fused[:, :512]
    cumsum_mean<<<BB * 2, 256, 0, stream>>>(h_buf, fused, seq_lens);

    // 6) final projection with bias + sequence mask
    gemm_bf16_wmma<<<dim3(BT/128, NPM/64), 256, 0, stream>>>(
        fused, H2, wmlp_p, H2, mlp_b, nullptr, out, nullptr, EE, 0, BT, EE, H2, seq_lens);

    (void)in_sizes; (void)n_in; (void)out_size; (void)ws_size;
}
